// EdgeUpdaterFrameDiff_44375602102682
// MI455X (gfx1250) — matrix operand order repaired and checked by reading, not compiled, not run on hardware
//
#include <hip/hip_runtime.h>

typedef __attribute__((ext_vector_type(16))) __bf16 v16bf;
typedef __attribute__((ext_vector_type(8)))  __bf16 v8bf;
typedef __attribute__((ext_vector_type(4)))  __bf16 v4bf;
typedef __attribute__((ext_vector_type(8)))  float  v8f;

#define L_DIM 512
#define ND    256     // NODE_DIM
#define NDH   128     // NODE_DIM/2
#define ED    128     // EDGE_DIM
#define MD    384     // MLP_DIM
#define MT    32      // edge rows per workgroup tile
#define KT    32      // K step per WMMA
#define NKS   (MD / KT)   // 12 K-steps
#define LN_EPS 1e-5f

// ---------------- helpers: async memory -> LDS path (CDNA5) ----------------
__device__ __forceinline__ unsigned lds_off_of(const void* p) {
  // generic pointers to LDS carry the wave-relative byte offset in the low 32b
  return (unsigned)(unsigned long long)p;
}

__device__ __forceinline__ void async_b128_to_lds(unsigned lvoff, unsigned gvoff,
                                                  unsigned long long gbase) {
  asm volatile("global_load_async_to_lds_b128 %0, %1, %2"
               :: "v"(lvoff), "v"(gvoff), "s"(gbase) : "memory");
}

__device__ __forceinline__ void wait_async0() {
  asm volatile("s_wait_asynccnt 0x0" ::: "memory");
}

// Stage a [rows x 32] bf16 weight K-slice (row-major [n][k]) into LDS.
// rows*4 chunks of 16B; each lane issues its own chunk -> async DMA, no VGPR data.
__device__ __forceinline__ void stage_wslice(unsigned lds_base,
                                             const __bf16* __restrict__ Wg,
                                             int k0, int rows, int tid) {
  const unsigned long long gbase =
      (unsigned long long)Wg + (unsigned)(k0 * 2);
  for (int c = tid; c < rows * 4; c += 256) {
    unsigned n = (unsigned)c >> 2;
    unsigned h = ((unsigned)c & 3) * 16;   // byte offset of 16B quarter-row
    async_b128_to_lds(lds_base + n * (KT * 2) + h, n * (MD * 2) + h, gbase);
  }
}

// ---------------- fp32 -> bf16 weight conversion (vectorized) ----------------
__global__ void cvt_bf16_kernel(const float4* __restrict__ src,
                                v4bf* __restrict__ dst, int n4) {
  int i = blockIdx.x * 256 + threadIdx.x;
  if (i < n4) {
    float4 f = src[i];
    v4bf p = { (__bf16)f.x, (__bf16)f.y, (__bf16)f.z, (__bf16)f.w };
    dst[i] = p;
  }
}

// ---------------- x_down = x @ Wd^T + bd   (512x128, bf16 out) ----------------
__global__ __launch_bounds__(256) void xdown_kernel(const float* __restrict__ x,
                                                    const float* __restrict__ Wd,
                                                    const float* __restrict__ bd,
                                                    __bf16* __restrict__ xdown) {
  const int wave = threadIdx.x >> 5;
  const int lane = threadIdx.x & 31;
  const int lo   = lane & 15;
  const int off  = (lane & 16) ? 8 : 0;
  const int mo   = off;
  const int m0   = blockIdx.x * 16;   // 32 blocks cover M=512
  const int n0   = wave * 16;         // 8 waves cover N=128

  v8f acc = {};
  const float* xrow = x  + (m0 + lo) * ND;
  const float* wrow = Wd + (n0 + lo) * ND;
  for (int k0 = 0; k0 < ND; k0 += KT) {
    v16bf a, b;
#pragma unroll
    for (int e = 0; e < 8; ++e) {
      a[e]     = (__bf16)xrow[k0 + off + e];
      a[e + 8] = (__bf16)xrow[k0 + 16 + off + e];
      b[e]     = (__bf16)wrow[k0 + off + e];
      b[e + 8] = (__bf16)wrow[k0 + 16 + off + e];
    }
    acc = __builtin_amdgcn_wmma_f32_16x16x32_bf16(false, a, false, b,
                                                  (short)0, acc, false, false);
  }
  const int   n  = n0 + lo;
  const float bv = bd[n];
#pragma unroll
  for (int r = 0; r < 8; ++r)
    xdown[(m0 + mo + r) * NDH + n] = (__bf16)(acc[r] + bv);
}

// ---------------- fused edge MLP: one 384x384 GEMM stage ----------------
// Double-buffered async weight staging overlapped with WMMA on the other buffer.
__device__ __forceinline__ void gemm384(const __bf16* s_src, __bf16* s_dst,
                                        __bf16* s_w0, __bf16* s_w1,
                                        const __bf16* __restrict__ Wg,
                                        const float* __restrict__ bias,
                                        const __bf16* s_res,  // residual or nullptr
                                        int tid, int wave, int lane) {
  const int lo  = lane & 15;
  const int off = (lane & 16) ? 8 : 0;
  const int mo  = off;
  const int m0  = (wave & 1) * 16;   // 2 m-tiles
  const int g   = wave >> 1;         // 4 groups x 6 n-tiles = N=384
  const unsigned wb0 = lds_off_of(s_w0);
  const unsigned wb1 = lds_off_of(s_w1);

  v8f acc[6] = {};

  stage_wslice(wb0, Wg, 0, MD, tid);               // prologue: slice 0 -> buf0
  for (int kk = 0; kk < NKS; ++kk) {
    const int cur = kk & 1;
    wait_async0();      // own async writes to buf(cur) landed
    __syncthreads();    // everyone's landed; everyone done reading buf(cur^1)
    if (kk + 1 < NKS)
      stage_wslice(cur ? wb0 : wb1, Wg, (kk + 1) * KT, MD, tid);

    const int k0 = kk * KT;
    v16bf a;
    {
      const __bf16* ap = &s_src[(m0 + lo) * MD + k0 + off];
      ((v8bf*)&a)[0] = *(const v8bf*)ap;
      ((v8bf*)&a)[1] = *(const v8bf*)(ap + 16);
    }
    const __bf16* sw = cur ? s_w1 : s_w0;
#pragma unroll
    for (int u = 0; u < 6; ++u) {
      v16bf b;
      const __bf16* bp = &sw[((g * 6 + u) * 16 + lo) * KT + off];
      ((v8bf*)&b)[0] = *(const v8bf*)bp;
      ((v8bf*)&b)[1] = *(const v8bf*)(bp + 16);
      acc[u] = __builtin_amdgcn_wmma_f32_16x16x32_bf16(false, a, false, b,
                                                       (short)0, acc[u],
                                                       false, false);
    }
  }
  // epilogue: bias + relu (+ residual), bf16 to LDS
#pragma unroll
  for (int u = 0; u < 6; ++u) {
    const int   n  = (g * 6 + u) * 16 + lo;
    const float bv = bias[n];
#pragma unroll
    for (int r = 0; r < 8; ++r) {
      float v = acc[u][r] + bv;
      v = v > 0.f ? v : 0.f;
      if (s_res) v += (float)s_res[(m0 + mo + r) * MD + n];
      s_dst[(m0 + mo + r) * MD + n] = (__bf16)v;
    }
  }
  __syncthreads();
}

// ---------------- main fused kernel ----------------
__global__ __launch_bounds__(256) void edge_mlp_kernel(
    const float* __restrict__ z, const __bf16* __restrict__ xdown,
    const __bf16* __restrict__ W1, const float* __restrict__ b1,
    const __bf16* __restrict__ W2, const float* __restrict__ b2,
    const __bf16* __restrict__ Wo, const float* __restrict__ bo,
    const float* __restrict__ gamma, const float* __restrict__ beta,
    float* __restrict__ out) {
  __shared__ __align__(16) __bf16 s_zin[MT * MD];   // 24 KB
  __shared__ __align__(16) __bf16 s_h1 [MT * MD];   // 24 KB
  __shared__ __align__(16) __bf16 s_h2 [MT * MD];   // 24 KB
  __shared__ __align__(16) __bf16 s_wa [MD * KT];   // 24 KB (weight buf 0)
  __shared__ __align__(16) __bf16 s_wb [MD * KT];   // 24 KB (weight buf 1)
  __shared__ __align__(16) float  s_out[MT * ED];   // 16 KB
  __shared__ float s_mu[MT], s_rs[MT];

  const int tid  = threadIdx.x;
  const int wave = tid >> 5;
  const int lane = tid & 31;
  const int lo   = lane & 15;
  const int off  = (lane & 16) ? 8 : 0;
  const int mo   = off;
  const int i    = blockIdx.x >> 4;          // 0..511
  const int j0   = (blockIdx.x & 15) * MT;   // 16 j-blocks of 32

  // ---- stage z_in = [xdown[i] | xdown[j] | bf16(z[i,j,:])] ----
  // node halves: 1024 x 16B copies
  for (int c = tid; c < MT * (ND / 8); c += 256) {
    int m = c >> 5, q = c & 31;              // q*8 = column in 0..255
    const __bf16* src = (q < 16) ? &xdown[i * NDH + q * 8]
                                 : &xdown[(j0 + m) * NDH + (q - 16) * 8];
    *(v8bf*)&s_zin[m * MD + q * 8] = *(const v8bf*)src;
  }
  // z slice: 1024 x float4 loads, packed bf16 stores
  for (int c = tid; c < MT * (ED / 4); c += 256) {
    int m = c >> 5, q = (c & 31) * 4;
    float4 f = *(const float4*)&z[((i * L_DIM) + (j0 + m)) * ED + q];
    v4bf p = { (__bf16)f.x, (__bf16)f.y, (__bf16)f.z, (__bf16)f.w };
    *(v4bf*)&s_zin[m * MD + ND + q] = p;
  }
  // (first wait+barrier inside gemm384 covers staging visibility)

  // h1 = relu(z_in @ W1^T + b1)
  gemm384(s_zin, s_h1, s_wa, s_wb, W1, b1, nullptr, tid, wave, lane);
  // h2 = relu(h1 @ W2^T + b2) + z_in  (residual fused)
  gemm384(s_h1, s_h2, s_wa, s_wb, W2, b2, s_zin, tid, wave, lane);

  // ---- out_pre = h2 @ Wo^T + bo  (N = 128; wave w owns n-tile w) ----
  {
    const unsigned wb0 = lds_off_of(s_wa);
    const unsigned wb1 = lds_off_of(s_wb);
    const int n0 = wave * 16;
    v8f acc[2] = {};
    stage_wslice(wb0, Wo, 0, ED, tid);
    for (int kk = 0; kk < NKS; ++kk) {
      const int cur = kk & 1;
      wait_async0();
      __syncthreads();
      if (kk + 1 < NKS)
        stage_wslice(cur ? wb0 : wb1, Wo, (kk + 1) * KT, ED, tid);

      const int k0 = kk * KT;
      const __bf16* sw = cur ? s_wb : s_wa;
      v16bf b;
      const __bf16* bp = &sw[(n0 + lo) * KT + off];
      ((v8bf*)&b)[0] = *(const v8bf*)bp;
      ((v8bf*)&b)[1] = *(const v8bf*)(bp + 16);
#pragma unroll
      for (int mt = 0; mt < 2; ++mt) {
        v16bf a;
        const __bf16* ap = &s_h2[(mt * 16 + lo) * MD + k0 + off];
        ((v8bf*)&a)[0] = *(const v8bf*)ap;
        ((v8bf*)&a)[1] = *(const v8bf*)(ap + 16);
        acc[mt] = __builtin_amdgcn_wmma_f32_16x16x32_bf16(false, a, false, b,
                                                          (short)0, acc[mt],
                                                          false, false);
      }
    }
    const int   n  = n0 + lo;
    const float bv = bo[n];
#pragma unroll
    for (int mt = 0; mt < 2; ++mt)
#pragma unroll
      for (int r = 0; r < 8; ++r)
        s_out[(mt * 16 + mo + r) * ED + n] = acc[mt][r] + bv;
  }
  __syncthreads();

  // ---- LayerNorm over last dim (128): 8 lanes per row, wave32 shuffles ----
  {
    const int row = tid >> 3;          // 32 rows
    const int sub = tid & 7;           // 8 lanes per row
    float s = 0.f, ss = 0.f;
    const float* rp = &s_out[row * ED + sub * 16];
#pragma unroll
    for (int c = 0; c < 16; ++c) {
      float v = rp[c];
      s += v; ss += v * v;
    }
#pragma unroll
    for (int m = 1; m < 8; m <<= 1) {
      s  += __shfl_xor(s, m, 32);
      ss += __shfl_xor(ss, m, 32);
    }
    if (sub == 0) {
      float mu  = s * (1.f / ED);
      float var = ss * (1.f / ED) - mu * mu;
      s_mu[row] = mu;
      s_rs[row] = rsqrtf(var + LN_EPS);
    }
  }
  __syncthreads();

  for (int t = tid; t < MT * (ED / 4); t += 256) {
    int m = t >> 5, q = (t & 31) * 4;
    float mu = s_mu[m], rs = s_rs[m];
    const float* sp = &s_out[m * ED + q];
    float4 o;
    o.x = (sp[0] - mu) * rs * gamma[q + 0] + beta[q + 0];
    o.y = (sp[1] - mu) * rs * gamma[q + 1] + beta[q + 1];
    o.z = (sp[2] - mu) * rs * gamma[q + 2] + beta[q + 2];
    o.w = (sp[3] - mu) * rs * gamma[q + 3] + beta[q + 3];
    *(float4*)&out[((i * L_DIM) + (j0 + m)) * ED + q] = o;
  }
}

// ---------------- host launch ----------------
extern "C" void kernel_launch(void* const* d_in, const int* in_sizes, int n_in,
                              void* d_out, int out_size, void* d_ws, size_t ws_size,
                              hipStream_t stream) {
  const float* x     = (const float*)d_in[0];
  const float* z     = (const float*)d_in[1];
  const float* Wd    = (const float*)d_in[2];
  const float* bd    = (const float*)d_in[3];
  const float* W1    = (const float*)d_in[4];
  const float* b1    = (const float*)d_in[5];
  const float* W2    = (const float*)d_in[6];
  const float* b2    = (const float*)d_in[7];
  const float* Wo    = (const float*)d_in[8];
  const float* bo    = (const float*)d_in[9];
  const float* gamma = (const float*)d_in[10];
  const float* beta  = (const float*)d_in[11];
  float* out = (float*)d_out;

  char* ws = (char*)d_ws;
  __bf16* W1b = (__bf16*)(ws + 0);              // 384*384*2 = 294912 B
  __bf16* W2b = (__bf16*)(ws + 294912);         // 294912 B
  __bf16* Wob = (__bf16*)(ws + 589824);         // 128*384*2 = 98304 B
  __bf16* xdb = (__bf16*)(ws + 688128);         // 512*128*2 = 131072 B

  cvt_bf16_kernel<<<(MD * MD / 4 + 255) / 256, 256, 0, stream>>>(
      (const float4*)W1, (v4bf*)W1b, MD * MD / 4);
  cvt_bf16_kernel<<<(MD * MD / 4 + 255) / 256, 256, 0, stream>>>(
      (const float4*)W2, (v4bf*)W2b, MD * MD / 4);
  cvt_bf16_kernel<<<(ED * MD / 4 + 255) / 256, 256, 0, stream>>>(
      (const float4*)Wo, (v4bf*)Wob, ED * MD / 4);

  xdown_kernel<<<L_DIM / 16, 256, 0, stream>>>(x, Wd, bd, xdb);

  edge_mlp_kernel<<<L_DIM * (L_DIM / MT), 256, 0, stream>>>(
      z, xdb, W1b, b1, W2b, b2, Wob, bo, gamma, beta, out);
}
